// MoELayer_32701880991798
// MI455X (gfx1250) — compile-verified, matrix operand order checked by reading, not run on hardware
//
#include <hip/hip_runtime.h>

// MoE dense layer:  out[t,:] = sum_e softmax(x@gw+gb)[t,e] * (x[t,:] @ W_e)
// Restructured as one f16 WMMA GEMM with effective K = E*D_IN, where the
// A-tile rows are scaled by the gate score of the expert owning the K block.

#define T_TOK 8192
#define D_IN  1024
#define D_OUT 1024
#define NEXP  8

#define BM 128
#define BN 128
#define BK 32
#define LDH 40                         // LDS row stride in halves (pad vs bank conflicts)
#define NSTEPS ((NEXP * D_IN) / BK)    // 256 K-steps

typedef __attribute__((ext_vector_type(16))) _Float16 v16h;
typedef __attribute__((ext_vector_type(8)))  float    v8f;

union FragU { v16h v; unsigned u[8]; };

// ---------------------------------------------------------------- gate ----
__global__ __launch_bounds__(256)
void moe_gate_kernel(const float* __restrict__ x,
                     const float* __restrict__ gw,
                     const float* __restrict__ gb,
                     float* __restrict__ gate)
{
    const int wave = threadIdx.x >> 5;        // wave32: one wave per token
    const int lane = threadIdx.x & 31;
    const int t    = blockIdx.x * 8 + wave;

    const float* xr = x + (size_t)t * D_IN;

    float acc[NEXP];
#pragma unroll
    for (int e = 0; e < NEXP; ++e) acc[e] = 0.f;

    for (int i = lane; i < D_IN; i += 32) {
        const float  xv = xr[i];
        const float4* wr = (const float4*)(gw + (size_t)i * NEXP);
        const float4 w0 = wr[0], w1 = wr[1];
        acc[0] += xv * w0.x; acc[1] += xv * w0.y;
        acc[2] += xv * w0.z; acc[3] += xv * w0.w;
        acc[4] += xv * w1.x; acc[5] += xv * w1.y;
        acc[6] += xv * w1.z; acc[7] += xv * w1.w;
    }
#pragma unroll
    for (int off = 16; off > 0; off >>= 1)
#pragma unroll
        for (int e = 0; e < NEXP; ++e)
            acc[e] += __shfl_xor(acc[e], off, 32);

    if (lane == 0) {
        float m = -3.4e38f;
#pragma unroll
        for (int e = 0; e < NEXP; ++e) { acc[e] += gb[e]; m = fmaxf(m, acc[e]); }
        float s = 0.f;
#pragma unroll
        for (int e = 0; e < NEXP; ++e) { acc[e] = __expf(acc[e] - m); s += acc[e]; }
        const float inv = 1.f / s;
#pragma unroll
        for (int e = 0; e < NEXP; ++e) gate[(size_t)t * NEXP + e] = acc[e] * inv;
    }
}

// ---------------------------------------------------------------- GEMM ----
__global__ __launch_bounds__(256)
void moe_gemm_kernel(const float* __restrict__ x,
                     const float* __restrict__ gate,
                     const float* __restrict__ ew,
                     float* __restrict__ out)
{
    __shared__ __align__(16) _Float16 Abuf[2][BM * LDH];   // [row][k]
    __shared__ __align__(16) _Float16 Bbuf[2][BN * LDH];   // [col][k] (transposed)

    const int tid    = threadIdx.x;
    const int lane   = tid & 31;
    const int laneLo = lane & 15;
    const int laneHi = lane >> 4;
    const int wave   = tid >> 5;
    const int wave_m = wave >> 1;           // 0..3, each owns 32 M rows
    const int wave_n = wave & 1;            // 0..1, each owns 64 N cols

    const int n0 = blockIdx.x * BN;
    const int t0 = blockIdx.y * BM;

    // staging task decomposition (fixed per thread)
    const int aRowBase = tid >> 3;          // +32*i  -> 128 rows
    const int aGrp     = tid & 7;           // 8 float4 groups per 32-float row
    const int bKBase   = tid >> 5;          // +8*i   -> 32 k rows
    const int bGrp     = tid & 31;          // 32 float4 groups per 128-float row

    const v8f vzero = {0.f, 0.f, 0.f, 0.f, 0.f, 0.f, 0.f, 0.f};
    v8f acc[2][4];
#pragma unroll
    for (int i = 0; i < 2; ++i)
#pragma unroll
        for (int j = 0; j < 4; ++j) acc[i][j] = vzero;

    float4 aReg[4]; float gReg[4]; float4 bReg[4];

    auto loadG = [&](int s) {
        const int e  = s >> 5;
        const int k0 = (s & 31) << 5;
        const float* xp = x + (size_t)t0 * D_IN + k0 + aGrp * 4;
#pragma unroll
        for (int i = 0; i < 4; ++i) {
            const int row = aRowBase + 32 * i;
            aReg[i] = *(const float4*)(xp + (size_t)row * D_IN);
            gReg[i] = gate[(size_t)(t0 + row) * NEXP + e];
        }
        const float* wp = ew + (size_t)e * D_IN * D_OUT
                             + (size_t)k0 * D_OUT + n0 + bGrp * 4;
#pragma unroll
        for (int i = 0; i < 4; ++i) {
            const int kr = bKBase + 8 * i;
            bReg[i] = *(const float4*)(wp + (size_t)kr * D_OUT);
        }
    };

    auto storeL = [&](int buf) {
#pragma unroll
        for (int i = 0; i < 4; ++i) {     // gate-scaled A rows, fp32 -> f16
            const int row = aRowBase + 32 * i;
            _Float16* d = &Abuf[buf][row * LDH + aGrp * 4];
            d[0] = (_Float16)(aReg[i].x * gReg[i]);
            d[1] = (_Float16)(aReg[i].y * gReg[i]);
            d[2] = (_Float16)(aReg[i].z * gReg[i]);
            d[3] = (_Float16)(aReg[i].w * gReg[i]);
        }
#pragma unroll
        for (int i = 0; i < 4; ++i) {     // B stored transposed: [col][k]
            const int kr  = bKBase + 8 * i;
            const int col = bGrp * 4;
            Bbuf[buf][(col + 0) * LDH + kr] = (_Float16)bReg[i].x;
            Bbuf[buf][(col + 1) * LDH + kr] = (_Float16)bReg[i].y;
            Bbuf[buf][(col + 2) * LDH + kr] = (_Float16)bReg[i].z;
            Bbuf[buf][(col + 3) * LDH + kr] = (_Float16)bReg[i].w;
        }
    };

    // 16-bit A/B fragment: element h=2r+j holds K = 16*(r>=4) + 8*laneHi + 2*(r&3) + j
    auto ldFrag = [&](const _Float16* base, int major) -> v16h {
        FragU f;
        const _Float16* p = base + major * LDH;
#pragma unroll
        for (int r = 0; r < 8; ++r) {
            const int k = ((r & 4) << 2) + (laneHi << 3) + ((r & 3) << 1);
            f.u[r] = *(const unsigned*)(p + k);
        }
        return f.v;
    };

    auto compute = [&](int buf) {
        v16h a[2], b[4];
#pragma unroll
        for (int tm = 0; tm < 2; ++tm)
            a[tm] = ldFrag(Abuf[buf], wave_m * 32 + tm * 16 + laneLo);
#pragma unroll
        for (int tn = 0; tn < 4; ++tn)
            b[tn] = ldFrag(Bbuf[buf], wave_n * 64 + tn * 16 + laneLo);
#pragma unroll
        for (int tm = 0; tm < 2; ++tm)
#pragma unroll
            for (int tn = 0; tn < 4; ++tn)
                acc[tm][tn] = __builtin_amdgcn_wmma_f32_16x16x32_f16(
                    false, a[tm], false, b[tn], (short)0, acc[tm][tn],
                    false, false);
    };

    // software pipeline: global->reg (s+1) | wmma (s) | reg->lds (s+1) | barrier
    loadG(0);
    storeL(0);
    __syncthreads();
    for (int s = 0; s < NSTEPS; ++s) {
        const int cur = s & 1;
        if (s + 1 < NSTEPS) loadG(s + 1);
        compute(cur);
        if (s + 1 < NSTEPS) storeL(cur ^ 1);
        __syncthreads();
    }

    // f32 C/D layout: VGPR r, lanes0-15 -> M=r ; lanes16-31 -> M=r+8 ; N=laneLo
#pragma unroll
    for (int tm = 0; tm < 2; ++tm)
#pragma unroll
        for (int tn = 0; tn < 4; ++tn) {
            const int Mb = t0 + wave_m * 32 + tm * 16 + laneHi * 8;
            const int Nc = n0 + wave_n * 64 + tn * 16 + laneLo;
#pragma unroll
            for (int r = 0; r < 8; ++r)
                out[(size_t)(Mb + r) * D_OUT + Nc] = acc[tm][tn][r];
        }
}

// --------------------------------------------------------------- launch ---
extern "C" void kernel_launch(void* const* d_in, const int* in_sizes, int n_in,
                              void* d_out, int out_size, void* d_ws, size_t ws_size,
                              hipStream_t stream)
{
    const float* x  = (const float*)d_in[0];   // [B,S,D_IN]
    const float* gw = (const float*)d_in[1];   // [D_IN,E]
    const float* gb = (const float*)d_in[2];   // [E]
    const float* ew = (const float*)d_in[3];   // [E,D_IN,D_OUT]
    float*       out  = (float*)d_out;         // [B,S,D_OUT]
    float*       gate = (float*)d_ws;          // [T_TOK,E] scratch (256 KB)

    moe_gate_kernel<<<T_TOK / 8, 256, 0, stream>>>(x, gw, gb, gate);

    dim3 grid(D_OUT / BN, T_TOK / BM);         // 8 x 64 blocks
    moe_gemm_kernel<<<grid, 256, 0, stream>>>(x, gate, ew, out);
}
